// MMDLayer_4861902979817
// MI455X (gfx1250) — matrix operand order repaired and checked by reading, not compiled, not run on hardware
//
#include <hip/hip_runtime.h>

typedef __attribute__((ext_vector_type(16))) _Float16 v16h;
typedef __attribute__((ext_vector_type(8)))  _Float16 v8h;
typedef __attribute__((ext_vector_type(4)))  _Float16 v4h;
typedef __attribute__((ext_vector_type(8)))  float    v8f;
typedef __attribute__((ext_vector_type(4)))  float    v4f;

// Problem constants (reference: B=1, T=4096, D=256, S=128, 18 outputs/step)
constexpr int Dk = 256;   // feature dim (= total WMMA K)
constexpr int Sb = 128;   // band width (shift-register length)

// ---------------------------------------------------------------------------
// Pass 1: build U in f16 once.  U[j] = d_init[127-j] for j < 128, else x[j-128].
// One thread converts 4 consecutive elements (f32x4 load -> f16x4 store).
// ---------------------------------------------------------------------------
__global__ __launch_bounds__(256)
void build_u_f16(const float* __restrict__ x, const float* __restrict__ d_init,
                 _Float16* __restrict__ U, int nrows) {
    const int e = (blockIdx.x * 256 + threadIdx.x) * 4;
    if (e >= nrows * Dk) return;
    const int j = e >> 8;        // row (Dk == 256)
    const int c = e & (Dk - 1);  // col
    const float* src = (j >= Sb) ? (x + (size_t)(j - Sb) * Dk + c)
                                 : (d_init + (size_t)(Sb - 1 - j) * Dk + c);
    const v4f f = *(const v4f*)src;
    v4h h;
    h.x = (_Float16)f.x; h.y = (_Float16)f.y;
    h.z = (_Float16)f.z; h.w = (_Float16)f.w;
    *(v4h*)(U + e) = h;
}

// ---------------------------------------------------------------------------
// Pass 2: one wave per 16-row t-tile of the banded Gram matrix
//   C[t, jj] = x[t] . U[jj],  jj in [t, t+127]   (A row t == U row 128+t)
// via v_wmma_f32_16x16x32_f16, then the (1+g)^3 prefix-sum epilogue.
// ---------------------------------------------------------------------------
__global__ __launch_bounds__(32)
void mmd_band_f16(const _Float16* __restrict__ U, float* __restrict__ out) {
    __shared__ float gbuf[16 * Sb];     // gamma band: [rowInTile][lag]

    const int lane = threadIdx.x;       // 0..31
    const int t0   = blockIdx.x * 16;   // t-tile base
    const int kh   = lane >> 4;         // K-block half select (0/1)
    const int l16  = lane & 15;         // A row / B column within tile

    // A tile: U rows 128+t0..128+t0+15, ISA 16-bit A(16x32) layout:
    // halves h=0..7 -> K = 32*kk + 8*kh + h ; h=8..15 -> K = +16
    v16h A[8];
    {
        const _Float16* ar = U + (size_t)(Sb + t0 + l16) * Dk;
        #pragma unroll
        for (int kk = 0; kk < 8; ++kk) {
            const _Float16* p = ar + 32 * kk + 8 * kh;
            const v8h lo = *(const v8h*)p;
            const v8h hi = *(const v8h*)(p + 16);
            A[kk] = __builtin_shufflevector(lo, hi, 0,1,2,3,4,5,6,7,
                                                    8,9,10,11,12,13,14,15);
        }
    }

    // 9 jj-tiles cover the band [t0, t0+143]
    for (int q = 0; q < 9; ++q) {
        const _Float16* br = U + (size_t)(t0 + 16 * q + l16) * Dk;
        v8f acc = {};
        #pragma unroll
        for (int kk = 0; kk < 8; ++kk) {
            // ISA 16-bit B(32x16) layout: halves h=0..7 -> K = 32*kk + 8*kh + h ; h=8..15 -> K = +16
            const _Float16* p = br + 32 * kk + 8 * kh;
            const v8h lo = *(const v8h*)p;
            const v8h hi = *(const v8h*)(p + 16);
            const v16h b = __builtin_shufflevector(lo, hi, 0,1,2,3,4,5,6,7,
                                                           8,9,10,11,12,13,14,15);
            acc = __builtin_amdgcn_wmma_f32_16x16x32_f16(
                      false, A[kk], false, b, (short)0, acc, false, false);
        }
        // C/D layout: VGPR r holds (M=r, lanes 0-15) / (M=r+8, lanes 16-31), N=l16.
        // Lag s = 127 + M - 16q - n; keep only the in-band part.
        #pragma unroll
        for (int r = 0; r < 8; ++r) {
            const int M = r + 8 * kh;
            const int s = (Sb - 1) + M - 16 * q - l16;
            if (s >= 0 && s < Sb)
                gbuf[M * Sb + s] = acc[r] * (1.0f / (float)Dk);
        }
    }

    __syncthreads();

    // Epilogue: poly = (1+g)^3; outputs 1..16 = 2*prefix, 17 = 2*total, 0 = 0.
    if (lane < 16) {
        float* orow = out + (size_t)(t0 + lane) * 18;
        float sum = 0.0f;
        orow[0] = 0.0f;
        #pragma unroll
        for (int s = 0; s < 16; ++s) {
            const float b = 1.0f + gbuf[lane * Sb + s];
            sum += b * b * b;
            orow[s + 1] = 2.0f * sum;
        }
        for (int s = 16; s < Sb; ++s) {
            const float b = 1.0f + gbuf[lane * Sb + s];
            sum += b * b * b;
        }
        orow[17] = 2.0f * sum;
    }
}

// ---------------------------------------------------------------------------
// Fallback: fused single-pass version (f32 loads + in-kernel conversion),
// used only if the workspace cannot hold the f16 U array.
// ---------------------------------------------------------------------------
__global__ __launch_bounds__(32)
void mmd_band_fused(const float* __restrict__ x,
                    const float* __restrict__ d_init,
                    float* __restrict__ out) {
    __shared__ float gbuf[16 * Sb];

    const int lane = threadIdx.x;
    const int t0   = blockIdx.x * 16;
    const int kh   = lane >> 4;
    const int l16  = lane & 15;

    v16h A[8];
    {
        const float* ar = x + (size_t)(t0 + l16) * Dk;
        #pragma unroll
        for (int kk = 0; kk < 8; ++kk) {
            const float* p = ar + 32 * kk + 8 * kh;
            v16h a;
            #pragma unroll
            for (int h = 0; h < 8; ++h) {
                a[h]     = (_Float16)p[h];
                a[h + 8] = (_Float16)p[16 + h];
            }
            A[kk] = a;
        }
    }

    for (int q = 0; q < 9; ++q) {
        const int jj = t0 + 16 * q + l16;
        const float* urow = (jj >= Sb) ? (x + (size_t)(jj - Sb) * Dk)
                                       : (d_init + (size_t)(Sb - 1 - jj) * Dk);
        v8f acc = {};
        #pragma unroll
        for (int kk = 0; kk < 8; ++kk) {
            const float* p = urow + 32 * kk + 8 * kh;
            v16h b;
            #pragma unroll
            for (int h = 0; h < 8; ++h) { b[h] = (_Float16)p[h]; b[h + 8] = (_Float16)p[16 + h]; }
            acc = __builtin_amdgcn_wmma_f32_16x16x32_f16(
                      false, A[kk], false, b, (short)0, acc, false, false);
        }
        #pragma unroll
        for (int r = 0; r < 8; ++r) {
            const int M = r + 8 * kh;
            const int s = (Sb - 1) + M - 16 * q - l16;
            if (s >= 0 && s < Sb)
                gbuf[M * Sb + s] = acc[r] * (1.0f / (float)Dk);
        }
    }

    __syncthreads();

    if (lane < 16) {
        float* orow = out + (size_t)(t0 + lane) * 18;
        float sum = 0.0f;
        orow[0] = 0.0f;
        #pragma unroll
        for (int s = 0; s < 16; ++s) {
            const float b = 1.0f + gbuf[lane * Sb + s];
            sum += b * b * b;
            orow[s + 1] = 2.0f * sum;
        }
        for (int s = 16; s < Sb; ++s) {
            const float b = 1.0f + gbuf[lane * Sb + s];
            sum += b * b * b;
        }
        orow[17] = 2.0f * sum;
    }
}

extern "C" void kernel_launch(void* const* d_in, const int* in_sizes, int n_in,
                              void* d_out, int out_size, void* d_ws, size_t ws_size,
                              hipStream_t stream) {
    const float* x      = (const float*)d_in[0];   // (1, T, 1, 256) f32
    const float* d_init = (const float*)d_in[1];   // (128, 256) f32
    float* out          = (float*)d_out;           // (1, T, 18) f32

    const int T      = in_sizes[0] / Dk;           // 4096
    const int nrows  = T + Sb;                     // U rows
    const size_t need = (size_t)nrows * Dk * sizeof(_Float16);

    if (ws_size >= need) {
        _Float16* U = (_Float16*)d_ws;
        const int nconv = (nrows * Dk) / 4;        // threads in pass 1
        build_u_f16<<<(nconv + 255) / 256, 256, 0, stream>>>(x, d_init, U, nrows);
        mmd_band_f16<<<T / 16, 32, 0, stream>>>(U, out);
    } else {
        mmd_band_fused<<<T / 16, 32, 0, stream>>>(x, d_init, out);
    }
}
